// VPDiffusionBlock_3685081940009
// MI455X (gfx1250) — matrix operand order repaired and checked by reading, not compiled, not run on hardware
//
#include <hip/hip_runtime.h>
#include <hip/hip_bf16.h>
#include <math.h>

// ---------------------------------------------------------------------------
// VPDiffusionBlock forward for MI455X (gfx1250, wave32, WMMA).
// GEMMs: v_wmma_f32_16x16x32_bf16, bf16 operands staged contiguously:
//   - weights pre-transposed to col-major bf16 (two contiguous 8-k runs per lane),
//   - activations mirrored in bf16 by producer kernels (2x16B loads + shuffle).
// Pair geometry is built as bf16 16x992 LDS tiles and fed to WMMA.
// pv path factored: sum_k att*(pair@Wpv) == (sum_k att*pair) @ Wpv.
// ---------------------------------------------------------------------------

#define NRES 2048
#define NA 5
#define NK 32
#define ND 256
#define NP 64
#define NB_BINS 16
#define NH 8
#define NHD 32
#define NRB 16
#define NSEGS 8
#define PAIRF 992           // 484 (pos) + 484 (prev) + 16 (dg) + 8 pad = 31*32
#define SIGMA_C 10.0f

typedef __bf16 bf16;
typedef bf16  v16bf __attribute__((ext_vector_type(16)));
typedef bf16  v8bf  __attribute__((ext_vector_type(8)));
typedef float v8f   __attribute__((ext_vector_type(8)));

__device__ __forceinline__ bf16 f2bf(float f) {
  unsigned u = __builtin_bit_cast(unsigned, f);
  unsigned r = u + 0x7FFFu + ((u >> 16) & 1u);   // round-to-nearest-even
  unsigned short h = (unsigned short)(r >> 16);
  return __builtin_bit_cast(bf16, h);
}
__device__ __forceinline__ float wsum(float v) {
#pragma unroll
  for (int o = 16; o > 0; o >>= 1) v += __shfl_xor(v, o, 32);
  return v;
}
__device__ __forceinline__ float wmaxr(float v) {
#pragma unroll
  for (int o = 16; o > 0; o >>= 1) v = fmaxf(v, __shfl_xor(v, o, 32));
  return v;
}
__device__ __forceinline__ float gelu_f(float x) {
  float x3 = x * x * x;
  return 0.5f * x * (1.0f + tanhf(0.7978845608028654f * (x + 0.044715f * x3)));
}
__device__ __forceinline__ v16bf cat8(v8bf lo, v8bf hi) {
  return __builtin_shufflevector(lo, hi, 0, 1, 2, 3, 4, 5, 6, 7, 8, 9, 10, 11,
                                 12, 13, 14, 15);
}

// ---------------------------------------------------------------------------
// Frames: R (N,3,3) columns = e1,e2,e3; ca; optional local_pos + point feats
// (feat written as bf16, padded 20 -> 32 with zeros: direct WMMA-A operand).
// ---------------------------------------------------------------------------
__global__ void k_frames(const float* __restrict__ pos, float* __restrict__ R,
                         float* __restrict__ ca, float* __restrict__ lp,
                         bf16* __restrict__ feat) {
  int n = blockIdx.x * blockDim.x + threadIdx.x;
  if (n >= NRES) return;
  const float* pn = pos + n * NA * 3;
  float Nx[3], Ca[3], Cx[3], e1[3], e2[3], e3[3], u[3];
  float s = 0.f;
#pragma unroll
  for (int c = 0; c < 3; c++) { Nx[c] = pn[c]; Ca[c] = pn[3 + c]; Cx[c] = pn[6 + c]; }
#pragma unroll
  for (int c = 0; c < 3; c++) { e1[c] = Cx[c] - Ca[c]; s += e1[c] * e1[c]; }
  float inv = 1.0f / sqrtf(fmaxf(s, 1e-12f));
#pragma unroll
  for (int c = 0; c < 3; c++) e1[c] *= inv;
  float d = 0.f;
#pragma unroll
  for (int c = 0; c < 3; c++) { u[c] = Nx[c] - Ca[c]; d += u[c] * e1[c]; }
  s = 0.f;
#pragma unroll
  for (int c = 0; c < 3; c++) { e2[c] = u[c] - d * e1[c]; s += e2[c] * e2[c]; }
  inv = 1.0f / sqrtf(fmaxf(s, 1e-12f));
#pragma unroll
  for (int c = 0; c < 3; c++) e2[c] *= inv;
  e3[0] = e1[1] * e2[2] - e1[2] * e2[1];
  e3[1] = e1[2] * e2[0] - e1[0] * e2[2];
  e3[2] = e1[0] * e2[1] - e1[1] * e2[0];
#pragma unroll
  for (int c = 0; c < 3; c++) {
    R[n * 9 + c * 3 + 0] = e1[c];
    R[n * 9 + c * 3 + 1] = e2[c];
    R[n * 9 + c * 3 + 2] = e3[c];
    ca[n * 3 + c] = Ca[c];
  }
  if (lp != nullptr) {
    float L[NA][3];
    for (int a = 0; a < NA; a++) {
      float px[3];
#pragma unroll
      for (int j = 0; j < 3; j++) px[j] = pn[a * 3 + j] - Ca[j];
      L[a][0] = px[0] * e1[0] + px[1] * e1[1] + px[2] * e1[2];
      L[a][1] = px[0] * e2[0] + px[1] * e2[1] + px[2] * e2[2];
      L[a][2] = px[0] * e3[0] + px[1] * e3[1] + px[2] * e3[2];
#pragma unroll
      for (int k = 0; k < 3; k++) lp[n * NA * 3 + a * 3 + k] = L[a][k];
    }
    float m[3] = {0.f, 0.f, 0.f};
    for (int a = 0; a < NA; a++)
      for (int k = 0; k < 3; k++) m[k] += L[a][k];
#pragma unroll
    for (int k = 0; k < 3; k++) m[k] *= (1.0f / NA);
    float q[NA][3];
    float mn = 0.f;
    for (int a = 0; a < NA; a++) {
      float ss = 0.f;
#pragma unroll
      for (int k = 0; k < 3; k++) { q[a][k] = L[a][k] - m[k]; ss += q[a][k] * q[a][k]; }
      mn += sqrtf(fmaxf(ss, 1e-12f));
    }
    mn *= (1.0f / NA);
    float invmn = 1.0f / mn;
    for (int a = 0; a < NA; a++) {
      float ss = 0.f;
#pragma unroll
      for (int k = 0; k < 3; k++) {
        float pv = q[a][k] * invmn;
        feat[n * 32 + a * 3 + k] = f2bf(pv);
        ss += pv * pv;
      }
      feat[n * 32 + 15 + a] = f2bf(sqrtf(fmaxf(ss, 1e-12f)));
    }
    for (int idx = 20; idx < 32; idx++) feat[n * 32 + idx] = f2bf(0.f);
  }
}

// ---------------------------------------------------------------------------
// Weight prep: Wt[col*KdPad + k] = bf16(W[k*Nout + col]), zero-padded.
// ---------------------------------------------------------------------------
__global__ void k_wprep(const float* __restrict__ W, bf16* __restrict__ Wt,
                        int Kd, int Nout, int KdPad, int NoutPad) {
  int i = blockIdx.x * blockDim.x + threadIdx.x;
  if (i >= NoutPad * KdPad) return;
  int col = i / KdPad, k = i % KdPad;
  float v = (col < Nout && k < Kd) ? W[k * Nout + col] : 0.f;
  Wt[i] = f2bf(v);
}

// ---------------------------------------------------------------------------
// WMMA GEMM: C[M,Nout] = act(A[M,KdPad](bf16) @ Wt(col-major bf16) + bias).
// One 16x32 output tile per wave; A frag = 2x contiguous 16B loads + shuffle,
// B frag = 2x contiguous 16B loads + shuffle per tile. KdPad % 32 == 0: no guards.
// ---------------------------------------------------------------------------
template <int ACT, int OUTBF>
__global__ void __launch_bounds__(256)
k_gemm(const bf16* __restrict__ A, const bf16* __restrict__ Wt,
       const float* __restrict__ bias, void* __restrict__ Cv,
       int M, int Kd, int Nout, int NoutPad) {
  int tilesN = (NoutPad + 31) >> 5;
  int tiles = (M >> 4) * tilesN;
  int wid = blockIdx.x * (blockDim.x >> 5) + (threadIdx.x >> 5);
  if (wid >= tiles) return;
  int tm = wid / tilesN, tn = wid % tilesN;
  int m0 = tm << 4, n0 = tn << 5;
  int lane = threadIdx.x & 31;
  int r15 = lane & 15, hi = lane >> 4;
  const bf16* Arow = A + (size_t)(m0 + r15) * Kd;
  int col0 = n0 + r15;
  int col1 = col0 + 16;
  bool has2 = (n0 + 16) < NoutPad;  // wave-uniform
  const bf16* B0 = Wt + (size_t)col0 * Kd;
  const bf16* B1 = Wt + (size_t)col1 * Kd;
  v8f acc0 = {0.f, 0.f, 0.f, 0.f, 0.f, 0.f, 0.f, 0.f};
  v8f acc1 = acc0;
  for (int k0 = 0; k0 < Kd; k0 += 32) {
    v8bf alo = *(const v8bf*)(Arow + k0 + (hi << 3));
    v8bf ahi = *(const v8bf*)(Arow + k0 + 16 + (hi << 3));
    v16bf a = cat8(alo, ahi);
    v16bf b0 = cat8(*(const v8bf*)(B0 + k0 + (hi << 3)), *(const v8bf*)(B0 + k0 + 16 + (hi << 3)));
    acc0 = __builtin_amdgcn_wmma_f32_16x16x32_bf16(false, a, false, b0, (short)0,
                                                   acc0, false, false);
    if (has2) {
      v16bf b1 = cat8(*(const v8bf*)(B1 + k0 + (hi << 3)), *(const v8bf*)(B1 + k0 + 16 + (hi << 3)));
      acc1 = __builtin_amdgcn_wmma_f32_16x16x32_bf16(false, a, false, b1, (short)0,
                                                     acc1, false, false);
    }
  }
  float* Cf = (float*)Cv;
  bf16* Cb = (bf16*)Cv;
  if (col0 < Nout) {
    float bs = bias ? bias[col0] : 0.f;
#pragma unroll
    for (int r = 0; r < 8; r++) {
      float v = acc0[r] + bs;
      if (ACT == 1) v = gelu_f(v);
      size_t idx = (size_t)(m0 + (hi << 3) + r) * Nout + col0;
      if (OUTBF) Cb[idx] = f2bf(v);
      else Cf[idx] = v;
    }
  }
  if (has2 && col1 < Nout) {
    float bs = bias ? bias[col1] : 0.f;
#pragma unroll
    for (int r = 0; r < 8; r++) {
      float v = acc1[r] + bs;
      if (ACT == 1) v = gelu_f(v);
      size_t idx = (size_t)(m0 + (hi << 3) + r) * Nout + col1;
      if (OUTBF) Cb[idx] = f2bf(v);
      else Cf[idx] = v;
    }
  }
}

// ---------------------------------------------------------------------------
// Fused residual LayerNorm over D=256 (wave per row) + incremental update.
// Emits f32 local (for residuals) and bf16 local (next GEMM operand).
// ---------------------------------------------------------------------------
__global__ void k_ln_res(const float* __restrict__ xin, const float* __restrict__ u,
                         const float* __restrict__ g, const float* __restrict__ b,
                         float* __restrict__ xout, bf16* __restrict__ xbf,
                         const float* __restrict__ incIn, float* __restrict__ incOut) {
  int row = blockIdx.x * (blockDim.x >> 5) + (threadIdx.x >> 5);
  if (row >= NRES) return;
  int lane = threadIdx.x & 31;
  float v[8];
  float s = 0.f;
#pragma unroll
  for (int j = 0; j < 8; j++) {
    int c = lane + 32 * j;
    v[j] = xin[row * ND + c] + u[row * ND + c];
    s += v[j];
  }
  s = wsum(s);
  float mean = s * (1.0f / ND);
  float t = 0.f;
#pragma unroll
  for (int j = 0; j < 8; j++) { float dd = v[j] - mean; t += dd * dd; }
  t = wsum(t);
  float rs = rsqrtf(t * (1.0f / ND) + 1e-5f);
#pragma unroll
  for (int j = 0; j < 8; j++) {
    int c = lane + 32 * j;
    float o = g[c] * (v[j] - mean) * rs + b[c];
    xout[row * ND + c] = o;
    xbf[row * ND + c] = f2bf(o);
    incOut[row * ND + c] = incIn[row * ND + c] + u[row * ND + c];
  }
}

// local_norm = local + LN(incremental)  (bf16 out; only feeds Wu GEMM)
__global__ void k_lnadd(const float* __restrict__ inc, const float* __restrict__ g,
                        const float* __restrict__ b, const float* __restrict__ loc,
                        bf16* __restrict__ out) {
  int row = blockIdx.x * (blockDim.x >> 5) + (threadIdx.x >> 5);
  if (row >= NRES) return;
  int lane = threadIdx.x & 31;
  float v[8];
  float s = 0.f;
#pragma unroll
  for (int j = 0; j < 8; j++) { v[j] = inc[row * ND + lane + 32 * j]; s += v[j]; }
  s = wsum(s);
  float mean = s * (1.0f / ND);
  float t = 0.f;
#pragma unroll
  for (int j = 0; j < 8; j++) { float dd = v[j] - mean; t += dd * dd; }
  t = wsum(t);
  float rs = rsqrtf(t * (1.0f / ND) + 1e-5f);
#pragma unroll
  for (int j = 0; j < 8; j++) {
    int c = lane + 32 * j;
    out[row * ND + c] = f2bf(loc[row * ND + c] + g[c] * (v[j] - mean) * rs + b[c]);
  }
}

__global__ void k_zero(float* __restrict__ p, int n) {
  int i = blockIdx.x * blockDim.x + threadIdx.x;
  if (i < n) p[i] = 0.f;
}

__global__ void k_segacc(const float* __restrict__ lu, const int* __restrict__ batch,
                         const int* __restrict__ chain,
                         const unsigned char* __restrict__ mask,
                         float* __restrict__ bsum, float* __restrict__ csum,
                         float* __restrict__ bcnt, float* __restrict__ ccnt) {
  int i = blockIdx.x * blockDim.x + threadIdx.x;
  if (i >= NRES * 512) return;
  int n = i >> 9, j = i & 511;
  float mf = mask[n] ? 1.f : 0.f;
  float v = lu[i] * mf;
  int bi = batch[n], ci = chain[n];
  if (bi >= 0 && bi < NSEGS) atomicAdd(&bsum[bi * 512 + j], v);
  if (ci >= 0 && ci < NSEGS) atomicAdd(&csum[ci * 512 + j], v);
  if (j == 0) {
    if (bi >= 0 && bi < NSEGS) atomicAdd(&bcnt[bi], mf);
    if (ci >= 0 && ci < NSEGS) atomicAdd(&ccnt[ci], mf);
  }
}

__global__ void k_gbuild(const int* __restrict__ batch, const int* __restrict__ chain,
                         const float* __restrict__ bsum, const float* __restrict__ csum,
                         const float* __restrict__ bcnt, const float* __restrict__ ccnt,
                         bf16* __restrict__ g) {
  int i = blockIdx.x * blockDim.x + threadIdx.x;
  if (i >= NRES * 512) return;
  int n = i >> 9, j = i & 511;
  int bi = batch[n], ci = chain[n];
  float vb = 0.f, vc = 0.f;
  if (bi >= 0 && bi < NSEGS) vb = bsum[bi * 512 + j] / fmaxf(bcnt[bi], 1e-6f);
  if (ci >= 0 && ci < NSEGS) vc = csum[ci * 512 + j] / fmaxf(ccnt[ci], 1e-6f);
  g[i] = f2bf(fmaxf(vb, 0.f) + fmaxf(vc, 0.f));
}

// Pair weights -> one bf16 col-major block [64 cols x PAIRF k].
__global__ void k_wcat(const float* __restrict__ Wd, const float* __restrict__ Wr,
                       const float* __restrict__ Wdir, const float* __restrict__ Wd2,
                       const float* __restrict__ Wr2, const float* __restrict__ Wdir2,
                       const float* __restrict__ Wdg, bf16* __restrict__ wcatT) {
  int i = blockIdx.x * blockDim.x + threadIdx.x;
  if (i >= PAIRF * NP) return;
  int c = i / PAIRF, r = i % PAIRF;
  float v = 0.f;
  if (r < 400) v = Wd[r * NP + c];
  else if (r < 409) v = Wr[(r - 400) * NP + c];
  else if (r < 484) v = Wdir[(r - 409) * NP + c];
  else if (r < 884) v = Wd2[(r - 484) * NP + c];
  else if (r < 893) v = Wr2[(r - 884) * NP + c];
  else if (r < 968) v = Wdir2[(r - 893) * NP + c];
  else if (r < 984) v = Wdg[(r - 968) * NP + c];
  wcatT[i] = f2bf(v);
}

// ---------------------------------------------------------------------------
// Pair kernel: 16 (n,k) pairs per block. Phase1: 16x992 bf16 features in LDS
// (RBF dists, rel-rotation, rotated dirs for pos & prev, distogram softmax).
// Phase2: 4 waves WMMA against col-major wcatT -> pair[16,64] (pre-LN).
// ---------------------------------------------------------------------------
__global__ void __launch_bounds__(128)
k_pair(const float* __restrict__ pos, const float* __restrict__ ppos,
       const float* __restrict__ R1, const float* __restrict__ R2,
       const int* __restrict__ nbr, const float* __restrict__ dgm,
       const bf16* __restrict__ wcatT, float* __restrict__ pairb) {
  __shared__ bf16 sf[16][PAIRF];
  int wave = threadIdx.x >> 5;
  int lane = threadIdx.x & 31;
  int base = blockIdx.x << 4;

  for (int t = 0; t < 4; t++) {
    int i = wave * 4 + t;
    int pi = base + i;
    int n = pi >> 5;
    int kk = pi & 31;
    int nbRaw = nbr[n * NK + kk];
    int nb = nbRaw & (NRES - 1);  // JAX-style wrap for -1
    for (int srcv = 0; srcv < 2; srcv++) {
      const float* P = srcv ? ppos : pos;
      const float* RR = srcv ? R2 : R1;
      int off = srcv * 484;
      if (lane < 25) {
        int ai = lane / 5, aj = lane % 5;
        float dx[3];
#pragma unroll
        for (int c = 0; c < 3; c++)
          dx[c] = P[nb * 15 + aj * 3 + c] - P[n * 15 + ai * 3 + c];
        float dist = sqrtf(fmaxf(dx[0] * dx[0] + dx[1] * dx[1] + dx[2] * dx[2], 1e-12f));
        const float sig = 22.0f / 16.0f;
        const float coef = -0.5f / (sig * sig);
#pragma unroll
        for (int r = 0; r < NRB; r++) {
          float cent = 22.0f * (float)r * (1.0f / 15.0f);
          float dd = dist - cent;
          sf[i][off + (ai * 5 + aj) * NRB + r] = f2bf(__expf(coef * dd * dd));
        }
        float idv = 1.0f / dist;
        float nd[3] = {dx[0] * idv, dx[1] * idv, dx[2] * idv};
#pragma unroll
        for (int j = 0; j < 3; j++) {
          float sv = nd[0] * RR[n * 9 + 0 + j] + nd[1] * RR[n * 9 + 3 + j] +
                     nd[2] * RR[n * 9 + 6 + j];
          sf[i][off + 409 + (ai * 5 + aj) * 3 + j] = f2bf(sv);
        }
      }
      if (lane < 9) {
        int bq = lane / 3, cq = lane % 3;
        float sv = 0.f;
#pragma unroll
        for (int a2 = 0; a2 < 3; a2++)
          sv += RR[n * 9 + a2 * 3 + bq] * RR[nb * 9 + a2 * 3 + cq];
        sf[i][off + 400 + lane] = f2bf(sv);
      }
    }
    if (lane < 16) {
      float v = dgm[((size_t)n * NRES + nb) * NB_BINS + lane];
      float m = v;
#pragma unroll
      for (int o = 8; o > 0; o >>= 1) m = fmaxf(m, __shfl_xor(m, o, 16));
      float e = __expf(v - m);
      float s = e;
#pragma unroll
      for (int o = 8; o > 0; o >>= 1) s += __shfl_xor(s, o, 16);
      sf[i][968 + lane] = f2bf(e / s);
    }
    if (lane >= 16 && lane < 24) sf[i][984 + lane - 16] = f2bf(0.f);
  }
  __syncthreads();

  int r15 = lane & 15, hi = lane >> 4;
  int col = (wave << 4) + r15;
  const bf16* Bc = wcatT + (size_t)col * PAIRF;
  v8f acc = {0.f, 0.f, 0.f, 0.f, 0.f, 0.f, 0.f, 0.f};
  for (int k0 = 0; k0 < PAIRF; k0 += 32) {
    v8bf alo = *(const v8bf*)&sf[r15][k0 + (hi << 3)];
    v8bf ahi = *(const v8bf*)&sf[r15][k0 + 16 + (hi << 3)];
    v16bf a = cat8(alo, ahi);
    v16bf b = cat8(*(const v8bf*)(Bc + k0 + (hi << 3)), *(const v8bf*)(Bc + k0 + 16 + (hi << 3)));
    acc = __builtin_amdgcn_wmma_f32_16x16x32_bf16(false, a, false, b, (short)0, acc,
                                                  false, false);
  }
#pragma unroll
  for (int r = 0; r < 8; r++) {
    int prow = base + (hi << 3) + r;
    pairb[(size_t)prow * NP + col] = acc[r];
  }
}

// Pair: + relpos one-hot row, LayerNorm over P=64, then mask-zero.
__global__ void k_pairln(float* __restrict__ pairb, const int* __restrict__ nbr,
                         const int* __restrict__ resi, const int* __restrict__ chain,
                         const int* __restrict__ batch,
                         const unsigned char* __restrict__ mask,
                         const float* __restrict__ Wrp, const float* __restrict__ g,
                         const float* __restrict__ b) {
  int row = blockIdx.x * (blockDim.x >> 5) + (threadIdx.x >> 5);
  if (row >= NRES * NK) return;
  int lane = threadIdx.x & 31;
  int n = row >> 5, kk = row & 31;
  int nbRaw = nbr[n * NK + kk];
  int nb = nbRaw & (NRES - 1);
  float same = (chain[nb] == chain[n] && batch[nb] == batch[n]) ? 1.f : 0.f;
  int dp = resi[nb] - resi[n];
  dp = dp < -32 ? -32 : (dp > 32 ? 32 : dp);
  dp += 32;
  float v[2];
  float s = 0.f;
#pragma unroll
  for (int j = 0; j < 2; j++) {
    int c = lane + 32 * j;
    v[j] = pairb[(size_t)row * NP + c] + same * Wrp[dp * NP + c];
    s += v[j];
  }
  s = wsum(s);
  float mean = s * (1.0f / NP);
  float t = 0.f;
#pragma unroll
  for (int j = 0; j < 2; j++) { float dd = v[j] - mean; t += dd * dd; }
  t = wsum(t);
  float rs = rsqrtf(t * (1.0f / NP) + 1e-5f);
  float pm = (mask[n] && mask[nb] && nbRaw != -1) ? 1.f : 0.f;
#pragma unroll
  for (int j = 0; j < 2; j++) {
    int c = lane + 32 * j;
    pairb[(size_t)row * NP + c] = pm * (g[c] * (v[j] - mean) * rs + b[c]);
  }
}

// Attention logits + softmax: one wave per (n,h), lane = neighbour k.
__global__ void k_att(const float* __restrict__ q, const float* __restrict__ kx,
                      const float* __restrict__ pairb, const float* __restrict__ Wb,
                      const int* __restrict__ nbr, const unsigned char* __restrict__ mask,
                      float* __restrict__ att) {
  int wid = blockIdx.x * (blockDim.x >> 5) + (threadIdx.x >> 5);
  if (wid >= NRES * NH) return;
  int lane = threadIdx.x & 31;
  int n = wid >> 3, h = wid & 7;
  int nbRaw = nbr[n * NK + lane];
  int nb = nbRaw & (NRES - 1);
  float lg = 0.f;
  const float* qp = q + n * ND + h * NHD;
  const float* kp = kx + nb * ND + h * NHD;
#pragma unroll
  for (int d2 = 0; d2 < NHD; d2++) lg += qp[d2] * kp[d2];
  lg *= 0.17677669529663687f;  // 1/sqrt(32)
  const float* pr = pairb + (size_t)(n * NK + lane) * NP;
#pragma unroll
  for (int p = 0; p < NP; p++) lg += pr[p] * Wb[p * NH + h];
  float pm = (mask[n] && mask[nb] && nbRaw != -1) ? 1.f : 0.f;
  if (!(pm > 0.f)) lg = -1e9f;
  float m = wmaxr(lg);
  float e = __expf(lg - m);
  float s = wsum(e);
  att[(size_t)wid * NK + lane] = e / s;
}

// wpair[n,h,:] = sum_k att[n,h,k] * pair[n,k,:]
__global__ void k_wpair(const float* __restrict__ att, const float* __restrict__ pairb,
                        float* __restrict__ wp) {
  int wid = blockIdx.x * (blockDim.x >> 5) + (threadIdx.x >> 5);
  if (wid >= NRES * NH) return;
  int lane = threadIdx.x & 31;
  int n = wid >> 3;
  float w0 = 0.f, w1 = 0.f;
  for (int k = 0; k < NK; k++) {
    float a = att[(size_t)wid * NK + k];
    const float* pr = pairb + (size_t)(n * NK + k) * NP;
    w0 += a * pr[lane];
    w1 += a * pr[lane + 32];
  }
  wp[(size_t)wid * NP + lane] = w0;
  wp[(size_t)wid * NP + lane + 32] = w1;
}

// o[n,h,d] = sum_k att*v[nb] + wpair[n,h,:] . Wpv[:,h*32+d]   (bf16 out)
__global__ void k_out(const float* __restrict__ att, const float* __restrict__ vx,
                      const float* __restrict__ wp, const float* __restrict__ Wpv,
                      const int* __restrict__ nbr, bf16* __restrict__ o) {
  int wid = blockIdx.x * (blockDim.x >> 5) + (threadIdx.x >> 5);
  if (wid >= NRES * NH) return;
  int lane = threadIdx.x & 31;
  int n = wid >> 3, h = wid & 7;
  float acc = 0.f;
  for (int k = 0; k < NK; k++) {
    int nb = nbr[n * NK + k] & (NRES - 1);
    acc += att[(size_t)wid * NK + k] * vx[nb * ND + h * NHD + lane];
  }
#pragma unroll
  for (int p = 0; p < NP; p++)
    acc += wp[(size_t)wid * NP + p] * Wpv[p * ND + h * NHD + lane];
  o[n * ND + h * NHD + lane] = f2bf(acc);
}

__global__ void k_mul(const float* __restrict__ h1, const float* __restrict__ h2,
                      bf16* __restrict__ out, int n) {
  int i = blockIdx.x * blockDim.x + threadIdx.x;
  if (i < n) out[i] = f2bf(h1[i] * h2[i]);
}

// pos_out[n,a,j] = R[n,j,:] . (local_pos + SIGMA*pos_update)[n,a,:] + ca[n,j]
__global__ void k_posout(const float* __restrict__ lp, const float* __restrict__ pu,
                         const float* __restrict__ R, const float* __restrict__ ca,
                         float* __restrict__ po) {
  int n = blockIdx.x * blockDim.x + threadIdx.x;
  if (n >= NRES) return;
  for (int a = 0; a < NA; a++) {
    float nl[3];
#pragma unroll
    for (int k = 0; k < 3; k++)
      nl[k] = lp[n * 15 + a * 3 + k] + SIGMA_C * pu[n * 15 + a * 3 + k];
#pragma unroll
    for (int j = 0; j < 3; j++)
      po[n * 15 + a * 3 + j] = R[n * 9 + j * 3 + 0] * nl[0] +
                               R[n * 9 + j * 3 + 1] * nl[1] +
                               R[n * 9 + j * 3 + 2] * nl[2] + ca[n * 3 + j];
  }
}

// ---------------------------------------------------------------------------
static inline void launch_gemm(int act, int outbf, const bf16* A, const bf16* Wt,
                               const float* bias, void* C, int M, int Kd, int Nout,
                               int NoutPad, hipStream_t s) {
  int tilesN = (NoutPad + 31) / 32;
  int tiles = (M / 16) * tilesN;
  int blocks = (tiles + 7) / 8;
  if (act == 1 && outbf == 1)
    k_gemm<1, 1><<<blocks, 256, 0, s>>>(A, Wt, bias, C, M, Kd, Nout, NoutPad);
  else if (act == 1)
    k_gemm<1, 0><<<blocks, 256, 0, s>>>(A, Wt, bias, C, M, Kd, Nout, NoutPad);
  else
    k_gemm<0, 0><<<blocks, 256, 0, s>>>(A, Wt, bias, C, M, Kd, Nout, NoutPad);
}

static inline void launch_wprep(const float* W, bf16* Wt, int Kd, int Nout,
                                int KdPad, int NoutPad, hipStream_t s) {
  int n = NoutPad * KdPad;
  k_wprep<<<(n + 255) / 256, 256, 0, s>>>(W, Wt, Kd, Nout, KdPad, NoutPad);
}

extern "C" void kernel_launch(void* const* d_in, const int* in_sizes, int n_in,
                              void* d_out, int out_size, void* d_ws, size_t ws_size,
                              hipStream_t stream) {
  (void)in_sizes; (void)n_in; (void)out_size; (void)ws_size;
  const float* in_local = (const float*)d_in[0];
  const float* in_incr  = (const float*)d_in[1];
  const float* in_pos   = (const float*)d_in[2];
  const float* in_ppos  = (const float*)d_in[3];
  const float* in_dgm   = (const float*)d_in[4];
  const int*   in_nbr   = (const int*)d_in[5];
  const int*   in_resi  = (const int*)d_in[6];
  const int*   in_chain = (const int*)d_in[7];
  const int*   in_batch = (const int*)d_in[8];
  const unsigned char* in_mask = (const unsigned char*)d_in[9];
  const float* W_p2l1 = (const float*)d_in[10];
  const float* W_p2l2 = (const float*)d_in[11];
  const float* Wg1 = (const float*)d_in[12];
  const float* bg1 = (const float*)d_in[13];
  const float* Wg2 = (const float*)d_in[14];
  const float* bg2 = (const float*)d_in[15];
  const float* Wd = (const float*)d_in[16];
  const float* Wr = (const float*)d_in[17];
  const float* Wdir = (const float*)d_in[18];
  const float* Wd2 = (const float*)d_in[19];
  const float* Wr2 = (const float*)d_in[20];
  const float* Wdir2 = (const float*)d_in[21];
  const float* Wdg = (const float*)d_in[22];
  const float* Wrp = (const float*)d_in[23];
  const float* lnp_g = (const float*)d_in[24];
  const float* lnp_b = (const float*)d_in[25];
  const float* Wq = (const float*)d_in[26];
  const float* Wk = (const float*)d_in[27];
  const float* Wv = (const float*)d_in[28];
  const float* Wb = (const float*)d_in[29];
  const float* Wpv = (const float*)d_in[30];
  const float* Wo = (const float*)d_in[31];
  const float* Wmh = (const float*)d_in[32];
  const float* Wmg = (const float*)d_in[33];
  const float* Wmo = (const float*)d_in[34];
  const float* ln1_g = (const float*)d_in[35];
  const float* ln1_b = (const float*)d_in[36];
  const float* ln2_g = (const float*)d_in[37];
  const float* ln2_b = (const float*)d_in[38];
  const float* ln3_g = (const float*)d_in[39];
  const float* ln3_b = (const float*)d_in[40];
  const float* ln4_g = (const float*)d_in[41];
  const float* ln4_b = (const float*)d_in[42];
  const float* lnf_g = (const float*)d_in[43];
  const float* lnf_b = (const float*)d_in[44];
  const float* Wu = (const float*)d_in[45];

  float* d_local = (float*)d_out;            // N*D
  float* d_incr = d_local + NRES * ND;       // N*D
  float* d_pos = d_incr + NRES * ND;         // N*A*3

  char* wsb = (char*)d_ws;
  size_t off = 0;
  auto alloc = [&](size_t bytes) -> void* {
    void* p = wsb + off;
    off = (off + bytes + 255) & ~(size_t)255;
    return p;
  };
  // f32 scratch
  float* R1 = (float*)alloc(NRES * 9 * 4);
  float* R2 = (float*)alloc(NRES * 9 * 4);
  float* ca1 = (float*)alloc(NRES * 3 * 4);
  float* ca2 = (float*)alloc(NRES * 3 * 4);
  float* lp = (float*)alloc(NRES * 15 * 4);
  float* hid1 = (float*)alloc((size_t)NRES * 1024 * 4);
  float* hid2 = (float*)alloc((size_t)NRES * 1024 * 4);
  float* lu = (float*)alloc((size_t)NRES * 512 * 4);
  float* ubuf = (float*)alloc((size_t)NRES * ND * 4);
  float* qb = (float*)alloc((size_t)NRES * ND * 4);
  float* kb = (float*)alloc((size_t)NRES * ND * 4);
  float* vb = (float*)alloc((size_t)NRES * ND * 4);
  float* pu = (float*)alloc((size_t)NRES * 15 * 4);
  float* pairb = (float*)alloc((size_t)NRES * NK * NP * 4);
  float* attb = (float*)alloc((size_t)NRES * NH * NK * 4);
  float* wpb = (float*)alloc((size_t)NRES * NH * NP * 4);
  float* segblk = (float*)alloc((NSEGS * 512 * 2 + NSEGS * 2) * 4);
  float* bsum = segblk;
  float* csum = segblk + NSEGS * 512;
  float* bcnt = csum + NSEGS * 512;
  float* ccnt = bcnt + NSEGS;
  // bf16 activation mirrors (WMMA A operands)
  bf16* featbf = (bf16*)alloc((size_t)NRES * 32 * 2);
  bf16* localbf = (bf16*)alloc((size_t)NRES * ND * 2);
  bf16* hid1bf = (bf16*)alloc((size_t)NRES * 1024 * 2);
  bf16* gbf = (bf16*)alloc((size_t)NRES * 512 * 2);
  bf16* mulbf = (bf16*)alloc((size_t)NRES * 1024 * 2);
  bf16* obf = (bf16*)alloc((size_t)NRES * ND * 2);
  bf16* lnormbf = (bf16*)alloc((size_t)NRES * ND * 2);
  // bf16 col-major weights (WMMA B operands)
  bf16* Wt_p2l1 = (bf16*)alloc((size_t)1024 * 32 * 2);
  bf16* Wt_p2l2 = (bf16*)alloc((size_t)256 * 1024 * 2);
  bf16* Wt_g1 = (bf16*)alloc((size_t)512 * 256 * 2);
  bf16* Wt_g2 = (bf16*)alloc((size_t)256 * 512 * 2);
  bf16* Wt_q = (bf16*)alloc((size_t)256 * 256 * 2);
  bf16* Wt_k = (bf16*)alloc((size_t)256 * 256 * 2);
  bf16* Wt_v = (bf16*)alloc((size_t)256 * 256 * 2);
  bf16* Wt_o = (bf16*)alloc((size_t)256 * 256 * 2);
  bf16* Wt_mg = (bf16*)alloc((size_t)1024 * 256 * 2);
  bf16* Wt_mh = (bf16*)alloc((size_t)1024 * 256 * 2);
  bf16* Wt_mo = (bf16*)alloc((size_t)256 * 1024 * 2);
  bf16* Wt_u = (bf16*)alloc((size_t)16 * 256 * 2);
  bf16* wcatT = (bf16*)alloc((size_t)NP * PAIRF * 2);

  // 0) weight prep (bf16, transposed, padded)
  launch_wprep(W_p2l1, Wt_p2l1, 20, 1024, 32, 1024, stream);
  launch_wprep(W_p2l2, Wt_p2l2, 1024, 256, 1024, 256, stream);
  launch_wprep(Wg1, Wt_g1, 256, 512, 256, 512, stream);
  launch_wprep(Wg2, Wt_g2, 512, 256, 512, 256, stream);
  launch_wprep(Wq, Wt_q, 256, 256, 256, 256, stream);
  launch_wprep(Wk, Wt_k, 256, 256, 256, 256, stream);
  launch_wprep(Wv, Wt_v, 256, 256, 256, 256, stream);
  launch_wprep(Wo, Wt_o, 256, 256, 256, 256, stream);
  launch_wprep(Wmg, Wt_mg, 256, 1024, 256, 1024, stream);
  launch_wprep(Wmh, Wt_mh, 256, 1024, 256, 1024, stream);
  launch_wprep(Wmo, Wt_mo, 1024, 256, 1024, 256, stream);
  launch_wprep(Wu, Wt_u, 256, 15, 256, 16, stream);
  k_wcat<<<(PAIRF * NP + 255) / 256, 256, 0, stream>>>(Wd, Wr, Wdir, Wd2, Wr2,
                                                       Wdir2, Wdg, wcatT);

  // 1) frames for pos (with local_pos + bf16 point feats) and prev_pos
  k_frames<<<NRES / 256, 256, 0, stream>>>(in_pos, R1, ca1, lp, featbf);
  k_frames<<<NRES / 256, 256, 0, stream>>>(in_ppos, R2, ca2, nullptr, nullptr);

  // 2) u1 = gelu(feat @ W_p2l1) @ W_p2l2 ; LN1 + incremental
  launch_gemm(1, 1, featbf, Wt_p2l1, nullptr, hid1bf, NRES, 32, 1024, 1024, stream);
  launch_gemm(0, 0, hid1bf, Wt_p2l2, nullptr, ubuf, NRES, 1024, 256, 256, stream);
  k_ln_res<<<NRES / 8, 256, 0, stream>>>(in_local, ubuf, ln1_g, ln1_b, d_local,
                                         localbf, in_incr, d_incr);

  // 3) segment-mean gate
  launch_gemm(0, 0, localbf, Wt_g1, bg1, lu, NRES, 256, 512, 512, stream);
  k_zero<<<(NSEGS * 1024 + NSEGS * 2 + 255) / 256, 256, 0, stream>>>(
      segblk, NSEGS * 1024 + NSEGS * 2);
  k_segacc<<<NRES * 512 / 256, 256, 0, stream>>>(lu, in_batch, in_chain, in_mask,
                                                 bsum, csum, bcnt, ccnt);
  k_gbuild<<<NRES * 512 / 256, 256, 0, stream>>>(in_batch, in_chain, bsum, csum,
                                                 bcnt, ccnt, gbf);
  launch_gemm(0, 0, gbf, Wt_g2, bg2, ubuf, NRES, 512, 256, 256, stream);
  k_ln_res<<<NRES / 8, 256, 0, stream>>>(d_local, ubuf, ln2_g, ln2_b, d_local,
                                         localbf, d_incr, d_incr);

  // 4) pair features -> WMMA -> LN(+relpos) + mask
  k_pair<<<NRES * NK / 16, 128, 0, stream>>>(in_pos, in_ppos, R1, R2, in_nbr,
                                             in_dgm, wcatT, pairb);
  k_pairln<<<NRES * NK / 8, 256, 0, stream>>>(pairb, in_nbr, in_resi, in_chain,
                                              in_batch, in_mask, Wrp, lnp_g, lnp_b);

  // 5) attention
  launch_gemm(0, 0, localbf, Wt_q, nullptr, qb, NRES, 256, 256, 256, stream);
  launch_gemm(0, 0, localbf, Wt_k, nullptr, kb, NRES, 256, 256, 256, stream);
  launch_gemm(0, 0, localbf, Wt_v, nullptr, vb, NRES, 256, 256, 256, stream);
  k_att<<<NRES * NH / 8, 256, 0, stream>>>(qb, kb, pairb, Wb, in_nbr, in_mask, attb);
  k_wpair<<<NRES * NH / 8, 256, 0, stream>>>(attb, pairb, wpb);
  k_out<<<NRES * NH / 8, 256, 0, stream>>>(attb, vb, wpb, Wpv, in_nbr, obf);
  launch_gemm(0, 0, obf, Wt_o, nullptr, ubuf, NRES, 256, 256, 256, stream);
  k_ln_res<<<NRES / 8, 256, 0, stream>>>(d_local, ubuf, ln3_g, ln3_b, d_local,
                                         localbf, d_incr, d_incr);

  // 6) gated MLP: u4 = (gelu(local@Wmg) * (local@Wmh)) @ Wmo
  launch_gemm(1, 0, localbf, Wt_mg, nullptr, hid1, NRES, 256, 1024, 1024, stream);
  launch_gemm(0, 0, localbf, Wt_mh, nullptr, hid2, NRES, 256, 1024, 1024, stream);
  k_mul<<<NRES * 1024 / 256, 256, 0, stream>>>(hid1, hid2, mulbf, NRES * 1024);
  launch_gemm(0, 0, mulbf, Wt_mo, nullptr, ubuf, NRES, 1024, 256, 256, stream);
  k_ln_res<<<NRES / 8, 256, 0, stream>>>(d_local, ubuf, ln4_g, ln4_b, d_local,
                                         localbf, d_incr, d_incr);

  // 7) position update
  k_lnadd<<<NRES / 8, 256, 0, stream>>>(d_incr, lnf_g, lnf_b, d_local, lnormbf);
  launch_gemm(0, 0, lnormbf, Wt_u, nullptr, pu, NRES, 256, 15, 16, stream);
  k_posout<<<NRES / 256, 256, 0, stream>>>(lp, pu, R1, ca1, d_pos);
}
